// TitansMemoryLayer_60335700574257
// MI455X (gfx1250) — matrix operand order repaired and checked by reading, not compiled, not run on hardware
//
#include <hip/hip_runtime.h>
#include <hip/hip_bf16.h>

// Problem constants (match reference)
#define DECAYF 0.9f
#define ONE_MINUS_DECAYF 0.1f
constexpr int Bb = 4, Ss = 8192, Dd = 1024, Mm = 1024;
constexpr int Nrows = Bb * Ss;                 // 32768
constexpr int CHUNK = 256, NCH = Ss / CHUNK;   // 32 chunks
constexpr float DECAY_POW_CHUNK = 1.93233476e-12f; // 0.9^256
constexpr int NPART = 256;

typedef __bf16 bf16_t;
typedef bf16_t v16bf __attribute__((ext_vector_type(16)));
typedef bf16_t v8bf  __attribute__((ext_vector_type(8)));
typedef float  v8f   __attribute__((ext_vector_type(8)));

union Frag { v16bf v; v8bf h[2]; };

static __device__ __forceinline__ unsigned short f2bf(float f) {
    unsigned u = __builtin_bit_cast(unsigned, f);
    u += 0x7FFFu + ((u >> 16) & 1u);   // round-to-nearest-even
    return (unsigned short)(u >> 16);
}

// ---------------------------------------------------------------- utilities
__global__ __launch_bounds__(256) void cvt_bf16_kernel(const float* __restrict__ src,
                                                       unsigned short* __restrict__ dst, int n) {
    int i = blockIdx.x * 256 + threadIdx.x;
    if (i < n) dst[i] = f2bf(src[i]);
}

__global__ void zero_f32_kernel(float* __restrict__ p, int n) {
    int i = blockIdx.x * blockDim.x + threadIdx.x;
    if (i < n) p[i] = 0.0f;
}

// ---------------------------------------------------------------- EMA scan
// Phase A: per-(b,chunk,d) local scan with h_in = 0.
__global__ __launch_bounds__(256) void scan_a_kernel(const float* __restrict__ x,
                                                     float* __restrict__ clocal) {
    int t = blockIdx.x * 256 + threadIdx.x;      // 0 .. B*NCH*D-1
    int d = t & (Dd - 1);
    int bc = t >> 10;                            // b*NCH + c
    int b = bc >> 5;                             // NCH == 32
    int c = bc & (NCH - 1);
    size_t base = ((size_t)(b * Ss + c * CHUNK)) * Dd + d;
    float h = 0.0f;
#pragma unroll 4
    for (int k = 0; k < CHUNK; ++k)
        h = DECAYF * h + ONE_MINUS_DECAYF * x[base + (size_t)k * Dd];
    clocal[t] = h;
}

// Phase B: serial carry combine across the 32 chunks per (b,d).
__global__ __launch_bounds__(256) void scan_b_kernel(const float* __restrict__ clocal,
                                                     float* __restrict__ carry) {
    int t = blockIdx.x * 256 + threadIdx.x;      // 0 .. B*D-1
    int d = t & (Dd - 1);
    int b = t >> 10;
    float h = 0.0f;
    for (int c = 0; c < NCH; ++c) {
        int idx = (b * NCH + c) * Dd + d;
        carry[idx] = h;                           // state entering this chunk
        h = DECAY_POW_CHUNK * h + clocal[idx];
    }
}

// Phase C: replay chunk from carry-in, emit bf16 states (state BEFORE consuming x[t]).
__global__ __launch_bounds__(256) void scan_c_kernel(const float* __restrict__ x,
                                                     const float* __restrict__ carry,
                                                     unsigned short* __restrict__ states) {
    int t = blockIdx.x * 256 + threadIdx.x;
    int d = t & (Dd - 1);
    int bc = t >> 10;
    int b = bc >> 5;
    int c = bc & (NCH - 1);
    size_t base = ((size_t)(b * Ss + c * CHUNK)) * Dd + d;
    float h = carry[t];
#pragma unroll 4
    for (int k = 0; k < CHUNK; ++k) {
        size_t idx = base + (size_t)k * Dd;
        states[idx] = f2bf(h);
        h = DECAYF * h + ONE_MINUS_DECAYF * x[idx];
    }
}

// ---------------------------------------------------------------- GEMM helpers
// A fragment: lane<16 holds row (lane), K {k0..k0+7, k0+16..k0+23};
//             lane>=16 holds row (lane-16), K {k0+8..k0+15, k0+24..k0+31}.
static __device__ __forceinline__ void load_a_frag(Frag& f, const unsigned short* p, int k0) {
    f.h[0] = *(const v8bf*)(p + k0);
    f.h[1] = *(const v8bf*)(p + k0 + 16);
}
// B fragment: lane<16 holds col (lane), K {k0..k0+7, k0+16..k0+23}; lane>=16 col (lane-16), K {k0+8..k0+15, k0+24..k0+31}.
static __device__ __forceinline__ void load_b_frag(Frag& f, const unsigned short* p, int k0) {
    f.h[0] = *(const v8bf*)(p + k0);
    f.h[1] = *(const v8bf*)(p + k0 + 16);
}

// 64x64 tile per wave: 4 A-frags x 4 B-frags -> 16 WMMAs per K=32 step.
// 32 FLOP per operand byte (2x the 32x32 version); ~210 VGPRs.

// ---------------------------------------------------------------- GEMM1: H = silu(states @ w1^T), bf16 out
__global__ __launch_bounds__(256) void gemm1_silu_kernel(const unsigned short* __restrict__ A,
                                                         const unsigned short* __restrict__ Bw,
                                                         unsigned short* __restrict__ H) {
    const int lane = threadIdx.x & 31;
    const int wave = threadIdx.x >> 5;
    const int tilesC = Mm / 64;                   // 16
    int tile = blockIdx.x * 8 + wave;
    int tr = tile / tilesC, tc = tile % tilesC;
    int row0 = tr * 64, col0 = tc * 64;
    const int K = Dd;

    const int half = lane >> 4;
    const int sub = lane & 15;
    const unsigned short* ap[4];
    const unsigned short* bp[4];
#pragma unroll
    for (int i = 0; i < 4; ++i) {
        ap[i] = A  + (size_t)(row0 + 16 * i + sub) * K + (half ? 8 : 0);
        bp[i] = Bw + (size_t)(col0 + 16 * i + sub) * K + (half ? 8 : 0);
    }

    v8f acc[4][4];
#pragma unroll
    for (int i = 0; i < 4; ++i)
#pragma unroll
        for (int j = 0; j < 4; ++j) acc[i][j] = v8f{};

    for (int k0 = 0; k0 < K; k0 += 32) {
        Frag af[4], bf[4];
#pragma unroll
        for (int i = 0; i < 4; ++i) load_a_frag(af[i], ap[i], k0);
#pragma unroll
        for (int j = 0; j < 4; ++j) load_b_frag(bf[j], bp[j], k0);
#pragma unroll
        for (int i = 0; i < 4; ++i)
#pragma unroll
            for (int j = 0; j < 4; ++j)
                acc[i][j] = __builtin_amdgcn_wmma_f32_16x16x32_bf16(
                    false, af[i].v, false, bf[j].v, (short)0, acc[i][j], false, false);
    }

    const int rOff = half * 8;
    const int cOff = sub;
#pragma unroll
    for (int i = 0; i < 4; ++i) {
#pragma unroll
        for (int j = 0; j < 4; ++j) {
#pragma unroll
            for (int e = 0; e < 8; ++e) {
                float a = acc[i][j][e];
                float s = a / (1.0f + __expf(-a));   // SiLU
                H[(size_t)(row0 + 16 * i + rOff + e) * Mm + (col0 + 16 * j + cOff)] = f2bf(s);
            }
        }
    }
}

// ---------------------------------------------------------------- GEMM2: out = h @ w2^T (fp32) + fused MSE partials
__global__ __launch_bounds__(256) void gemm2_loss_kernel(const unsigned short* __restrict__ A,
                                                         const unsigned short* __restrict__ Bw,
                                                         const float* __restrict__ X,
                                                         float* __restrict__ Out,
                                                         float* __restrict__ partials) {
    const int lane = threadIdx.x & 31;
    const int wave = threadIdx.x >> 5;
    const int tilesC = Dd / 64;                   // 16
    int tile = blockIdx.x * 8 + wave;
    int tr = tile / tilesC, tc = tile % tilesC;
    int row0 = tr * 64, col0 = tc * 64;
    const int K = Mm;

    const int half = lane >> 4;
    const int sub = lane & 15;
    const unsigned short* ap[4];
    const unsigned short* bp[4];
#pragma unroll
    for (int i = 0; i < 4; ++i) {
        ap[i] = A  + (size_t)(row0 + 16 * i + sub) * K + (half ? 8 : 0);
        bp[i] = Bw + (size_t)(col0 + 16 * i + sub) * K + (half ? 8 : 0);
    }

    v8f acc[4][4];
#pragma unroll
    for (int i = 0; i < 4; ++i)
#pragma unroll
        for (int j = 0; j < 4; ++j) acc[i][j] = v8f{};

    for (int k0 = 0; k0 < K; k0 += 32) {
        Frag af[4], bf[4];
#pragma unroll
        for (int i = 0; i < 4; ++i) load_a_frag(af[i], ap[i], k0);
#pragma unroll
        for (int j = 0; j < 4; ++j) load_b_frag(bf[j], bp[j], k0);
#pragma unroll
        for (int i = 0; i < 4; ++i)
#pragma unroll
            for (int j = 0; j < 4; ++j)
                acc[i][j] = __builtin_amdgcn_wmma_f32_16x16x32_bf16(
                    false, af[i].v, false, bf[j].v, (short)0, acc[i][j], false, false);
    }

    const int rOff = half * 8;
    const int cOff = sub;
    float lsum = 0.0f;
#pragma unroll
    for (int i = 0; i < 4; ++i) {
#pragma unroll
        for (int j = 0; j < 4; ++j) {
#pragma unroll
            for (int e = 0; e < 8; ++e) {
                size_t idx = (size_t)(row0 + 16 * i + rOff + e) * Dd + (col0 + 16 * j + cOff);
                float a = acc[i][j][e];
                Out[idx] = a;
                float dxv = a - X[idx];
                lsum += dxv * dxv;
            }
        }
    }

    // wave32 reduce, then per-block reduce, then spread atomics over NPART slots
    for (int off = 16; off > 0; off >>= 1) lsum += __shfl_down(lsum, off);
    __shared__ float smem[8];
    if (lane == 0) smem[wave] = lsum;
    __syncthreads();
    if (threadIdx.x == 0) {
        float t = 0.0f;
#pragma unroll
        for (int i = 0; i < 8; ++i) t += smem[i];
        atomicAdd(&partials[blockIdx.x & (NPART - 1)], t);
    }
}

__global__ __launch_bounds__(256) void finalize_loss_kernel(const float* __restrict__ partials,
                                                            float* __restrict__ out_loss) {
    const int lane = threadIdx.x & 31;
    const int wave = threadIdx.x >> 5;
    float v = partials[threadIdx.x];
    for (int off = 16; off > 0; off >>= 1) v += __shfl_down(v, off);
    __shared__ float smem[8];
    if (lane == 0) smem[wave] = v;
    __syncthreads();
    if (threadIdx.x == 0) {
        float t = 0.0f;
#pragma unroll
        for (int i = 0; i < 8; ++i) t += smem[i];
        *out_loss = t * (1.0f / (float)((size_t)Nrows * Dd));
    }
}

// ---------------------------------------------------------------- launch
extern "C" void kernel_launch(void* const* d_in, const int* in_sizes, int n_in,
                              void* d_out, int out_size, void* d_ws, size_t ws_size,
                              hipStream_t stream) {
    (void)in_sizes; (void)n_in; (void)out_size; (void)ws_size;
    const float* x  = (const float*)d_in[0];   // [B,S,D]
    const float* w1 = (const float*)d_in[1];   // [M,D]
    const float* w2 = (const float*)d_in[2];   // [D,M]
    float* out = (float*)d_out;                // [B,S,D] + 1 (loss)

    char* ws = (char*)d_ws;
    size_t off = 0;
    auto wsalloc = [&](size_t bytes) -> void* {
        void* p = ws + off;
        off += (bytes + 255) & ~(size_t)255;
        return p;
    };
    unsigned short* states = (unsigned short*)wsalloc((size_t)Nrows * Dd * 2);  // 64 MB bf16
    unsigned short* hbuf   = (unsigned short*)wsalloc((size_t)Nrows * Mm * 2);  // 64 MB bf16
    unsigned short* w1b    = (unsigned short*)wsalloc((size_t)Mm * Dd * 2);     //  2 MB
    unsigned short* w2b    = (unsigned short*)wsalloc((size_t)Dd * Mm * 2);     //  2 MB
    float* clocal          = (float*)wsalloc((size_t)Bb * NCH * Dd * 4);        // 512 KB
    float* carry           = (float*)wsalloc((size_t)Bb * NCH * Dd * 4);        // 512 KB
    float* partials        = (float*)wsalloc(NPART * 4);

    cvt_bf16_kernel<<<(Mm * Dd + 255) / 256, 256, 0, stream>>>(w1, w1b, Mm * Dd);
    cvt_bf16_kernel<<<(Dd * Mm + 255) / 256, 256, 0, stream>>>(w2, w2b, Dd * Mm);
    zero_f32_kernel<<<1, NPART, 0, stream>>>(partials, NPART);

    scan_a_kernel<<<(Bb * NCH * Dd) / 256, 256, 0, stream>>>(x, clocal);
    scan_b_kernel<<<(Bb * Dd) / 256, 256, 0, stream>>>(clocal, carry);
    scan_c_kernel<<<(Bb * NCH * Dd) / 256, 256, 0, stream>>>(x, carry, states);

    // 64x64 tile per wave, 8 waves per block
    int tiles1 = (Nrows / 64) * (Mm / 64);        // 8192 waves
    gemm1_silu_kernel<<<tiles1 / 8, 256, 0, stream>>>(states, w1b, hbuf);
    int tiles2 = (Nrows / 64) * (Dd / 64);        // 8192 waves
    gemm2_loss_kernel<<<tiles2 / 8, 256, 0, stream>>>(hbuf, w2b, x, out, partials);

    finalize_loss_kernel<<<1, NPART, 0, stream>>>(partials, out + (size_t)Nrows * Dd);
}
